// LearnedRayModel_44152263803437
// MI455X (gfx1250) — compile-verified, matrix operand order checked by reading, not run on hardware
//
#include <hip/hip_runtime.h>
#include <math.h>

typedef __attribute__((ext_vector_type(16))) _Float16 v16h;
typedef __attribute__((ext_vector_type(8)))  float    v8f;
typedef __attribute__((ext_vector_type(4)))  float    f4;

constexpr int   kBrays  = 1048576;
constexpr int   kS      = 4;
constexpr int   kH      = 2048;
constexpr int   kW      = 2048;
constexpr int   kC      = 4;
constexpr int   kHid    = 32;
constexpr int   kOut    = 3;
constexpr float kEps    = 1e-5f;
constexpr float kPi     = 3.14159265358979323846f;
constexpr int   kTiles  = kBrays / 16;   // 65536 tiles of 16 rays
constexpr int   kBlocks = 4096;          // grid-stride: 16 tiles per wave

// K index held by (half-register h, half-wave hi) in the 16-bit WMMA A layout:
// lanes 0-15: h0..7 -> K0..7, h8..15 -> K16..23 ; lanes 16-31: +8
__device__ __forceinline__ int k_of(int h, int hi) {
    return (h & 7) + ((h & 8) ? 16 : 0) + (hi ? 8 : 0);
}

// Build one B operand (KxN row-major weight, column `col`), zero-padded rows/cols.
__device__ __forceinline__ v16h load_b_operand(const float* __restrict__ w,
                                               int ncols, int col, int hi, int kmax) {
    v16h b;
#pragma unroll
    for (int h = 0; h < 16; ++h) {
        const int k = k_of(h, hi);
        const float v = (col >= 0 && k < kmax) ? w[k * ncols + col] : 0.0f;
        b[h] = (_Float16)v;
    }
    return b;
}

__device__ __forceinline__ f4 sample_bilinear(const float* __restrict__ tex,
                                              float t, float p) {
    const float x  = t * (float)(kW - 1);
    const float y  = p * (float)(kH - 1);
    const float xf0 = floorf(x), yf0 = floorf(y);
    const int x0 = (int)xf0,        y0 = (int)yf0;
    const int x1 = (int)ceilf(x),   y1 = (int)ceilf(y);
    const float xf = x - xf0, yf = y - yf0;
    const f4* t4 = (const f4*)tex;
    const f4 c00 = t4[y0 * kW + x0];
    const f4 c01 = t4[y0 * kW + x1];
    const f4 c10 = t4[y1 * kW + x0];
    const f4 c11 = t4[y1 * kW + x1];
    const f4 top = c00 + xf * (c01 - c00);
    const f4 bot = c10 + xf * (c11 - c10);
    return top + yf * (bot - top);
}

__global__ void __launch_bounds__(32)
learned_ray_mlp(const float* __restrict__ inputs,
                const float* __restrict__ tex,
                const float* __restrict__ w_in,  const float* __restrict__ b_in,
                const float* __restrict__ g_in,  const float* __restrict__ bt_in,
                const float* __restrict__ w_hid, const float* __restrict__ b_hid,
                const float* __restrict__ g_hid, const float* __restrict__ bt_hid,
                const float* __restrict__ w_out, const float* __restrict__ b_out,
                float* __restrict__ out)
{
    __shared__ float act[16 * kHid];   // 16 rays x 32 hidden (f32 transpose buffer)
    __shared__ float ln_g[3 * kHid];
    __shared__ float ln_b[3 * kHid];

    const int lane = threadIdx.x;      // wave32
    const int hi   = lane >> 4;        // half-wave id (K split / M split)
    const int n    = lane & 15;        // column id (D layout) == ray slot (A layout)

    // Layernorm params -> LDS (layer 0: *_in ; layers 1,2: *_hid)
    ln_g[lane]      = g_in[lane];        ln_b[lane]      = bt_in[lane];
    ln_g[32 + lane] = g_hid[lane];       ln_b[32 + lane] = bt_hid[lane];
    ln_g[64 + lane] = g_hid[32 + lane];  ln_b[64 + lane] = bt_hid[32 + lane];

    // B operands in registers, reused across the tile loop.
    v16h Bw[3][2];
    Bw[0][0] = load_b_operand(w_in,               kHid, n,      hi, 16);
    Bw[0][1] = load_b_operand(w_in,               kHid, 16 + n, hi, 16);
    Bw[1][0] = load_b_operand(w_hid,              kHid, n,      hi, 32);
    Bw[1][1] = load_b_operand(w_hid,              kHid, 16 + n, hi, 32);
    Bw[2][0] = load_b_operand(w_hid + kHid*kHid,  kHid, n,      hi, 32);
    Bw[2][1] = load_b_operand(w_hid + kHid*kHid,  kHid, 16 + n, hi, 32);
    const v16h Bo = load_b_operand(w_out, kOut, (n < kOut) ? n : -1, hi, 32);

    const float bias_lo[3] = { b_in[n],      b_hid[n],      b_hid[32 + n]      };
    const float bias_hi[3] = { b_in[16 + n], b_hid[16 + n], b_hid[32 + 16 + n] };
    const float bias_out   = (n < kOut) ? b_out[n] : 0.0f;

    const f4* __restrict__ in4 = (const f4*)inputs;

    __syncthreads();

    for (int tile = blockIdx.x; tile < kTiles; tile += kBlocks) {
        const int ray = tile * 16 + n;             // A-matrix row M = n
        v16h A;
#pragma unroll
        for (int h = 0; h < 16; ++h) A[h] = (_Float16)0.0f;

        // One coalesced, non-temporal b128 load gives this lane's two sphere
        // angle pairs: inputs[ray*8 + 4*hi .. +3] (streamed once -> keep out of L2).
        const f4 ang = __builtin_nontemporal_load(&in4[ray * 2 + hi]);

        // Gather: half-wave hi samples spheres {2*hi, 2*hi+1} -> features K = s*4+c,
        // which is exactly this lane's K range in the WMMA A layout.
#pragma unroll
        for (int sl = 0; sl < 2; ++sl) {
            const int s = 2 * hi + sl;
            const float th = sl ? ang.z : ang.x;
            const float ph = sl ? ang.w : ang.y;
            float t = th * (0.5f / kPi);
            float p = ph * (1.0f / kPi);
            const bool fin = __builtin_isfinite(t) && __builtin_isfinite(p);
            t = fin ? t : 0.0f;
            p = fin ? p : 0.0f;
            const f4 f = sample_bilinear(tex + (size_t)s * kH * kW * kC, t, p);
            const float m = fin ? 1.0f : 0.0f;
            A[sl * 4 + 0] = (_Float16)(f.x * m);
            A[sl * 4 + 1] = (_Float16)(f.y * m);
            A[sl * 4 + 2] = (_Float16)(f.z * m);
            A[sl * 4 + 3] = (_Float16)(f.w * m);
        }

        // Three layers of relu(layernorm(A @ W + b))
#pragma unroll
        for (int ly = 0; ly < 3; ++ly) {
            v8f c0, c1;                            // bias folded into accumulator
#pragma unroll
            for (int r = 0; r < 8; ++r) { c0[r] = bias_lo[ly]; c1[r] = bias_hi[ly]; }
            c0 = __builtin_amdgcn_wmma_f32_16x16x32_f16(false, A, false, Bw[ly][0],
                                                        (short)0, c0, false, false);
            c1 = __builtin_amdgcn_wmma_f32_16x16x32_f16(false, A, false, Bw[ly][1],
                                                        (short)0, c1, false, false);
            // D layout -> LDS: VGPR r is ray (r + 8*hi), lane gives hidden col n / n+16
#pragma unroll
            for (int r = 0; r < 8; ++r) {
                const int m = r + 8 * hi;
                act[m * kHid + n]      = c0[r];
                act[m * kHid + 16 + n] = c1[r];
            }
            __syncthreads();                        // single-wave WG: S_NOP + fence
            // Read back in A layout; per-ray layernorm stats via one xor-16 shuffle
            float xv[16];
            float s1 = 0.0f, s2 = 0.0f;
#pragma unroll
            for (int h = 0; h < 16; ++h) {
                const int k = k_of(h, hi);
                const float v = act[n * kHid + k];
                xv[h] = v; s1 += v; s2 += v * v;
            }
            s1 += __shfl_xor(s1, 16, 32);
            s2 += __shfl_xor(s2, 16, 32);
            const float mean = s1 * (1.0f / kHid);
            const float var  = s2 * (1.0f / kHid) - mean * mean;
            const float rstd = rsqrtf(var + kEps);
#pragma unroll
            for (int h = 0; h < 16; ++h) {
                const int k = k_of(h, hi);
                const float y = (xv[h] - mean) * rstd * ln_g[ly * kHid + k]
                                + ln_b[ly * kHid + k];
                A[h] = (_Float16)fmaxf(y, 0.0f);
            }
            __syncthreads();                        // act reused next layer / tile
        }

        // Output layer: (16 x 32) @ (32 x 3), N padded to 16
        v8f co;
#pragma unroll
        for (int r = 0; r < 8; ++r) co[r] = bias_out;
        co = __builtin_amdgcn_wmma_f32_16x16x32_f16(false, A, false, Bo,
                                                    (short)0, co, false, false);
        if (n < kOut) {
#pragma unroll
            for (int r = 0; r < 8; ++r) {
                const int m = r + 8 * hi;
                // streamed once -> non-temporal store, keep L2 for textures
                __builtin_nontemporal_store(co[r],
                    &out[(size_t)(tile * 16 + m) * kOut + n]);
            }
        }
    }
}

extern "C" void kernel_launch(void* const* d_in, const int* in_sizes, int n_in,
                              void* d_out, int out_size, void* d_ws, size_t ws_size,
                              hipStream_t stream) {
    const float* inputs = (const float*)d_in[0];
    const float* tex    = (const float*)d_in[1];
    const float* w_in   = (const float*)d_in[2];
    const float* b_in   = (const float*)d_in[3];
    const float* g_in   = (const float*)d_in[4];
    const float* bt_in  = (const float*)d_in[5];
    const float* w_hid  = (const float*)d_in[6];
    const float* b_hid  = (const float*)d_in[7];
    const float* g_hid  = (const float*)d_in[8];
    const float* bt_hid = (const float*)d_in[9];
    const float* w_out  = (const float*)d_in[10];
    const float* b_out  = (const float*)d_in[11];
    float* out = (float*)d_out;

    learned_ray_mlp<<<dim3(kBlocks), dim3(32), 0, stream>>>(
        inputs, tex, w_in, b_in, g_in, bt_in,
        w_hid, b_hid, g_hid, bt_hid, w_out, b_out, out);
}
